// Custom_U_2_88630945120527
// MI455X (gfx1250) — matrix operand order repaired and checked by reading, not compiled, not run on hardware
//
#include <hip/hip_runtime.h>
#include <hip/hip_bf16.h>

// ---------------------------------------------------------------------------
// Problem constants (from the reference): L=2048, B=8, D_IN=1024, P=512,
// D_OUT=1024.  M = L*B = 16384 rows everywhere (x is [L,B,D] contiguous, so
// flattening [L,B] to M keeps row order identical to the reference).
// ---------------------------------------------------------------------------
#define LL   2048
#define BB   8
#define DIN  1024
#define PP   512
#define DOUT 1024
#define MM   (LL * BB)   // 16384

// gfx1250 async global->LDS copy (ASYNCcnt-tracked), with safe fallback.
#if defined(__has_builtin)
#if __has_builtin(__builtin_amdgcn_global_load_async_to_lds_b128) && \
    __has_builtin(__builtin_amdgcn_s_wait_asynccnt)
#define USE_ASYNC_LDS 1
#endif
#endif
#ifndef USE_ASYNC_LDS
#define USE_ASYNC_LDS 0
#endif

typedef __bf16 bf16_t;
typedef __attribute__((ext_vector_type(16))) __bf16 v16bf;
typedef __attribute__((ext_vector_type(8)))  __bf16 v8bf;
typedef __attribute__((ext_vector_type(8)))  float  v8f;

// Exact parameter types of the async-copy builtin (from hipcc diagnostics):
// param0 = v4i in AS1 (global), param1 = v4i in AS3 (LDS), then 2 int imms.
typedef int v4i_b __attribute__((vector_size(16)));
typedef __attribute__((address_space(1))) v4i_b* as1_v4i_p;
typedef __attribute__((address_space(3))) v4i_b* as3_v4i_p;

__device__ __forceinline__ __bf16 f2bf(float f) {
  unsigned u = __builtin_bit_cast(unsigned, f);
  u += 0x7FFFu + ((u >> 16) & 1u);          // round-to-nearest-even
  unsigned short h = (unsigned short)(u >> 16);
  return __builtin_bit_cast(__bf16, h);
}

__device__ __forceinline__ v8f wmma_bf16(v16bf a, v16bf b, v8f c) {
  // D(f32 16x16) = A(bf16 16x32) * B(bf16 32x16) + C
  return __builtin_amdgcn_wmma_f32_16x16x32_bf16(
      /*neg_a=*/false, a, /*neg_b=*/false, b,
      /*c_mod=*/(short)0, c, /*reuse_a=*/false, /*reuse_b=*/false);
}

__device__ __forceinline__ v16bf cat8(v8bf lo, v8bf hi) {
  return __builtin_shufflevector(lo, hi, 0,1,2,3,4,5,6,7,8,9,10,11,12,13,14,15);
}

// ---------------------------------------------------------------------------
// f32 -> bf16 elementwise convert
// ---------------------------------------------------------------------------
__global__ void k_f32_to_bf16(const float* __restrict__ in,
                              __bf16* __restrict__ out, int n) {
  int i = blockIdx.x * blockDim.x + threadIdx.x;
  if (i < n) out[i] = f2bf(in[i]);
}

// ---------------------------------------------------------------------------
// Build vT[b][p][l] = V[l][b][p] = kv[(l*B+b)*2P + P + p]  (bf16)
// so attention's P~@V B-operand (column of V = row of vT) is contiguous.
// ---------------------------------------------------------------------------
__global__ void k_build_vT(const __bf16* __restrict__ kv,
                           __bf16* __restrict__ vT) {
  long idx = (long)blockIdx.x * blockDim.x + threadIdx.x; // over B*P*L
  const long total = (long)BB * PP * LL;
  if (idx >= total) return;
  int l = (int)(idx % LL);
  long t = idx / LL;
  int p = (int)(t % PP);
  int b = (int)(t / PP);
  vT[idx] = kv[((long)l * BB + b) * (2 * PP) + PP + p];
}

// ---------------------------------------------------------------------------
// bf16 GEMM: C[M,N] = A[M,K] * Bm[N,K]^T (torch Linear layout: weight [out,in])
// Block = 256 threads = 8 waves; each wave owns a 16x64 tile (1 A-tile reused
// across 4 B-tiles -> 4 WMMA per K-step). Block tile = 128x64. B operands are
// L2-resident weights, so they stay on the direct global path.
// ---------------------------------------------------------------------------
template <bool WF32, bool WBF16>
__global__ __launch_bounds__(256) void k_gemm_bf16(
    const __bf16* __restrict__ A, const __bf16* __restrict__ Bm,
    float* __restrict__ Cf, __bf16* __restrict__ Cb, int M, int N, int K) {
  const int lane  = threadIdx.x & 31;
  const int wave  = threadIdx.x >> 5;
  const int lm    = lane & 15;
  const int hi    = lane >> 4;           // half-lane group
  const int koffA = hi ? 16 : 0;         // A-layout K split (contiguous, matches B)
  const int koffB = hi ? 16 : 0;         // B-layout K split
  const int m0 = blockIdx.y * 128 + wave * 16;
  const int n0 = blockIdx.x * 64;

  v8f acc[4] = {};
  const __bf16* aRow = A + (size_t)(m0 + lm) * K;

  for (int k = 0; k < K; k += 32) {
    __builtin_prefetch(aRow + k + 128, 0, 1);     // global_prefetch_b8
    v8bf alo = *(const v8bf*)(aRow + k + koffA);
    v8bf ahi = *(const v8bf*)(aRow + k + 8 + koffA);
    v16bf a = cat8(alo, ahi);
#pragma unroll
    for (int cc = 0; cc < 4; ++cc) {
      const __bf16* bRow = Bm + (size_t)(n0 + cc * 16 + lm) * K + k + koffB;
      v16bf b = *(const v16bf*)bRow;
      acc[cc] = wmma_bf16(a, b, acc[cc]);
    }
  }

  const int rbase = m0 + hi * 8;        // C-layout: VGPR r -> row rbase+r
#pragma unroll
  for (int cc = 0; cc < 4; ++cc) {
    const int col = n0 + cc * 16 + lm;
#pragma unroll
    for (int r = 0; r < 8; ++r) {
      size_t o = (size_t)(rbase + r) * N + col;
      if (WF32)  Cf[o] = acc[cc][r];
      if (WBF16) Cb[o] = f2bf(acc[cc][r]);
    }
  }
}

// ---------------------------------------------------------------------------
// Causal flash attention, one 16-row query tile per block, batch = blockIdx.y.
// The 32-key K panel (32x512 bf16 = 32KB) is shared by all 8 waves, so it is
// staged once per block into LDS via gfx1250 async global->LDS copies
// (ASYNCcnt), then consumed as WMMA B operands with conflict-free
// ds_load_b128 (row stride padded to 520 elements).  8 waves each own a
// 64-wide V-column slice; S and softmax stats are wave-private (redundant S
// compute avoids any cross-wave softmax exchange).  Online softmax matches
// the reference's additive -10000 causal mask.  Epilogue adds the f32
// residual (z) and emits bf16 for the final GEMM.
// ---------------------------------------------------------------------------
#define KSTRIDE 520   // 512 + 8 pad: row shift = 4 dword-banks -> no conflicts

__global__ __launch_bounds__(256) void k_attention(
    const __bf16* __restrict__ zb,   // [M, P]  (Q source; scale folded in S)
    const __bf16* __restrict__ kvb,  // [M, 2P] (K = cols 0..P)
    const __bf16* __restrict__ vT,   // [B, P, L]
    const float*  __restrict__ zf,   // [M, P]  residual
    __bf16* __restrict__ ao)         // [M, P]  attn-out + residual (bf16)
{
  __shared__ __align__(16) __bf16 kTile[32 * KSTRIDE];   // 33,280 B
  __shared__ __align__(16) __bf16 pTile[8][16 * 32];     //  8,192 B

  const int tid   = threadIdx.x;
  const int lane  = tid & 31;
  const int wave  = tid >> 5;
  const int lm    = lane & 15;
  const int hi    = lane >> 4;
  const int koffA = hi ? 16 : 0;
  const int koffB = hi ? 16 : 0;

  const int q0 = blockIdx.x * 16;      // query rows q0..q0+15
  const int b  = blockIdx.y;
  const int c0 = wave * 64;            // this wave's V-column slice

  const float scaling = 0.044194173824159216f;  // P^-0.5

  float m[8], lsum[8];
#pragma unroll
  for (int r = 0; r < 8; ++r) { m[r] = -3.0e38f; lsum[r] = 0.0f; }
  v8f o[4] = {};

  const __bf16* qRow = zb + (size_t)((q0 + lm) * BB + b) * PP;
  __bf16* pw = pTile[wave];

  const int jmax = q0 + 15;            // last key tile on/below the diagonal
  for (int j = 0; j <= jmax; j += 32) {
    // ---- stage 32-key K panel into LDS (2048 x 16B chunks / 256 thr) ----
#pragma unroll
    for (int c = 0; c < 8; ++c) {
      const int chunk = tid + 256 * c;         // coalesced: tid-consecutive
      const int key   = chunk >> 6;            // 0..31
      const int seg   = chunk & 63;            // 16B segment within 1KB row
      const __bf16* gp = kvb + ((size_t)(j + key) * BB + b) * (2 * PP) + seg * 8;
      __bf16* lp = kTile + (size_t)key * KSTRIDE + seg * 8;
#if USE_ASYNC_LDS
      void* gpv = (void*)gp;                   // strip const (generic)
      __builtin_amdgcn_global_load_async_to_lds_b128(
          (as1_v4i_p)gpv, (as3_v4i_p)(void*)lp, 0, 0);
#else
      *(v8bf*)lp = *(const v8bf*)gp;
#endif
    }
#if USE_ASYNC_LDS
    __builtin_amdgcn_s_wait_asynccnt(0);
#endif
    __syncthreads();

    // ---- S = Q @ K^T for 32 keys (two 16x16 C tiles), K-loop over P ----
    v8f s0a = {}, s1a = {};
    for (int p = 0; p < PP; p += 32) {
      v8bf alo = *(const v8bf*)(qRow + p + koffA);
      v8bf ahi = *(const v8bf*)(qRow + p + 8 + koffA);
      v16bf a = cat8(alo, ahi);
      const __bf16* kr0 = kTile + (size_t)lm * KSTRIDE + p + koffB;
      const __bf16* kr1 = kTile + (size_t)(16 + lm) * KSTRIDE + p + koffB;
      v16bf b0 = cat8(*(const v8bf*)kr0, *(const v8bf*)(kr0 + 8));
      v16bf b1 = cat8(*(const v8bf*)kr1, *(const v8bf*)(kr1 + 8));
      s0a = wmma_bf16(a, b0, s0a);
      s1a = wmma_bf16(a, b1, s1a);
    }

    // ---- scale, causal mask, online softmax (row = slot r in my half) ----
    float p0v[8], p1v[8], alpha[8];
#pragma unroll
    for (int r = 0; r < 8; ++r) {
      const int qr   = q0 + hi * 8 + r;
      const int key0 = j + lm, key1 = j + 16 + lm;
      float s0 = s0a[r] * scaling + (key0 > qr ? -10000.0f : 0.0f);
      float s1 = s1a[r] * scaling + (key1 > qr ? -10000.0f : 0.0f);
      float tm = fmaxf(s0, s1);
      tm = fmaxf(tm, __shfl_xor(tm, 1));
      tm = fmaxf(tm, __shfl_xor(tm, 2));
      tm = fmaxf(tm, __shfl_xor(tm, 4));
      tm = fmaxf(tm, __shfl_xor(tm, 8));     // max over all 32 keys of row qr
      float mn = fmaxf(m[r], tm);
      float al = __expf(m[r] - mn);
      float e0 = __expf(s0 - mn);
      float e1 = __expf(s1 - mn);
      float ts = e0 + e1;
      ts += __shfl_xor(ts, 1);
      ts += __shfl_xor(ts, 2);
      ts += __shfl_xor(ts, 4);
      ts += __shfl_xor(ts, 8);
      lsum[r] = lsum[r] * al + ts;
      m[r] = mn; alpha[r] = al; p0v[r] = e0; p1v[r] = e1;
    }

    // rescale running output
#pragma unroll
    for (int cc = 0; cc < 4; ++cc)
#pragma unroll
      for (int r = 0; r < 8; ++r) o[cc][r] *= alpha[r];

    // ---- C-layout -> A-layout re-tile of P~ via wave-private LDS ----
#pragma unroll
    for (int r = 0; r < 8; ++r) {
      pw[(hi * 8 + r) * 32 + lm]      = f2bf(p0v[r]);
      pw[(hi * 8 + r) * 32 + 16 + lm] = f2bf(p1v[r]);
    }
    v8bf plo = *(const v8bf*)(pw + lm * 32 + koffA);
    v8bf phi = *(const v8bf*)(pw + lm * 32 + 8 + koffA);
    v16bf pa = cat8(plo, phi);

    // ---- O += P~(16x32) @ V(32 x 64-slice), B from transposed vT ----
#pragma unroll
    for (int cc = 0; cc < 4; ++cc) {
      const __bf16* vp =
          vT + ((size_t)b * PP + (c0 + cc * 16 + lm)) * LL + j + koffB;
      v16bf bv = *(const v16bf*)vp;
      o[cc] = wmma_bf16(pa, bv, o[cc]);
    }
    __syncthreads();   // kTile WAR before next panel overwrite
  }

  // ---- epilogue: O / l  + residual, emit bf16 ----
#pragma unroll
  for (int cc = 0; cc < 4; ++cc) {
    const int col = c0 + cc * 16 + lm;
#pragma unroll
    for (int r = 0; r < 8; ++r) {
      const int qr = q0 + hi * 8 + r;
      size_t idx = (size_t)(qr * BB + b) * PP + col;
      float val = o[cc][r] / lsum[r] + zf[idx];
      ao[idx] = f2bf(val);
    }
  }
}

// ---------------------------------------------------------------------------
// Workspace layout (bytes, 1MB-multiple offsets; total ~147 MB)
// ---------------------------------------------------------------------------
#define OFF_XB   ((size_t)0)                         // x bf16      32 MB
#define OFF_W1B  ((size_t)33554432)                  // W1 bf16      1 MB
#define OFF_W2B  ((size_t)34603008)                  // W2 bf16      1 MB
#define OFF_W3B  ((size_t)35651584)                  // W3 bf16      1 MB
#define OFF_ZF   ((size_t)36700160)                  // z f32       32 MB
#define OFF_ZB   ((size_t)70254592)                  // z bf16      16 MB
#define OFF_KVB  ((size_t)87031808)                  // kv bf16     32 MB
#define OFF_VT   ((size_t)120586240)                 // vT bf16     16 MB
#define OFF_AO   ((size_t)137363456)                 // attn+res    16 MB

extern "C" void kernel_launch(void* const* d_in, const int* in_sizes, int n_in,
                              void* d_out, int out_size, void* d_ws,
                              size_t ws_size, hipStream_t stream) {
  (void)in_sizes; (void)n_in; (void)out_size; (void)ws_size;
  const float* x  = (const float*)d_in[0];
  const float* W1 = (const float*)d_in[1];
  const float* W2 = (const float*)d_in[2];
  const float* W3 = (const float*)d_in[3];
  float* out = (float*)d_out;
  char* ws = (char*)d_ws;

  __bf16* xb  = (__bf16*)(ws + OFF_XB);
  __bf16* w1b = (__bf16*)(ws + OFF_W1B);
  __bf16* w2b = (__bf16*)(ws + OFF_W2B);
  __bf16* w3b = (__bf16*)(ws + OFF_W3B);
  float*  zf  = (float*)(ws + OFF_ZF);
  __bf16* zb  = (__bf16*)(ws + OFF_ZB);
  __bf16* kvb = (__bf16*)(ws + OFF_KVB);
  __bf16* vtb = (__bf16*)(ws + OFF_VT);
  __bf16* aob = (__bf16*)(ws + OFF_AO);

  // 1) f32 -> bf16 conversions
  {
    int n = MM * DIN;
    k_f32_to_bf16<<<(n + 255) / 256, 256, 0, stream>>>(x, xb, n);
    n = PP * DIN;
    k_f32_to_bf16<<<(n + 255) / 256, 256, 0, stream>>>(W1, w1b, n);
    n = 2 * PP * PP;
    k_f32_to_bf16<<<(n + 255) / 256, 256, 0, stream>>>(W2, w2b, n);
    n = DOUT * PP;
    k_f32_to_bf16<<<(n + 255) / 256, 256, 0, stream>>>(W3, w3b, n);
  }

  // 2) z = x @ W1^T   [16384,512]  (f32 residual + bf16 feed)
  k_gemm_bf16<true, true><<<dim3(PP / 64, MM / 128), 256, 0, stream>>>(
      xb, w1b, zf, zb, MM, PP, DIN);

  // 3) kv = z @ W2^T  [16384,1024] (bf16) ; then vT[b][p][l] for P~@V
  k_gemm_bf16<false, true><<<dim3((2 * PP) / 64, MM / 128), 256, 0, stream>>>(
      zb, w2b, nullptr, kvb, MM, 2 * PP, PP);
  {
    long n = (long)BB * PP * LL;
    k_build_vT<<<(unsigned)((n + 255) / 256), 256, 0, stream>>>(kvb, vtb);
  }

  // 4) causal flash attention + residual -> bf16
  k_attention<<<dim3(LL / 16, BB), 256, 0, stream>>>(zb, kvb, vtb, zf, aob);

  // 5) d_out = (attn + residual) @ W3^T  [16384,1024] f32
  k_gemm_bf16<true, false><<<dim3(DOUT / 64, MM / 128), 256, 0, stream>>>(
      aob, w3b, out, nullptr, MM, DOUT, PP);
}
